// WindowAttention3D_v1_63247688401319
// MI455X (gfx1250) — compile-verified, matrix operand order checked by reading, not run on hardware
//
#include <hip/hip_runtime.h>
#include <hip/hip_bf16.h>

typedef __attribute__((ext_vector_type(16))) _Float16 v16h;
typedef __attribute__((ext_vector_type(8)))  float    v8f;

// ---------------------------------------------------------------------------
// Fragment loaders for V_WMMA_F32_16X16X32_F16 (wave32).
// A (16x32, f16): lane l -> row = l&15, kb = (l<16)?0:8;
//   VGPR j holds K = kb + 2j + (j>=4 ? 8 : 0), and K+1  (two contiguous
//   8-half runs per lane -> compiler emits 2x b128 loads per fragment).
// B (32x16, f16): same lane/K striping with lane -> column.
// C/D (16x16, f32): VGPR r -> row r (lanes 0-15) / row r+8 (lanes 16-31),
//   column = lane & 15.
// ---------------------------------------------------------------------------

__device__ inline v16h frag_rowmajor(const _Float16* __restrict__ base, int ld, int lane) {
  const int row = lane & 15;
  const int kb  = (lane < 16) ? 0 : 8;
  v16h f;
#pragma unroll
  for (int j = 0; j < 8; ++j) {
    const int k = kb + 2 * j + (j >= 4 ? 8 : 0);
    f[2 * j]     = base[row * ld + k];
    f[2 * j + 1] = base[row * ld + k + 1];
  }
  return f;
}

// Load B fragment from a K-major source (element [k][col] at base[k*ld + col]).
__device__ inline v16h frag_kmajor(const _Float16* __restrict__ base, int ld, int lane) {
  const int col = lane & 15;
  const int kb  = (lane < 16) ? 0 : 8;
  v16h f;
#pragma unroll
  for (int j = 0; j < 8; ++j) {
    const int k = kb + 2 * j + (j >= 4 ? 8 : 0);
    f[2 * j]     = base[k * ld + col];
    f[2 * j + 1] = base[(k + 1) * ld + col];
  }
  return f;
}

__device__ inline v8f wmma_f16(v16h a, v16h b, v8f c) {
  return __builtin_amdgcn_wmma_f32_16x16x32_f16(false, a, false, b, (short)0, c,
                                                false, false);
}

// ---------------------------------------------------------------------------
// Kernel 0: fp32 -> f16 weight conversion
// ---------------------------------------------------------------------------
__global__ void cvt_f32_f16(const float* __restrict__ src,
                            _Float16* __restrict__ dst, int n) {
  int i = blockIdx.x * 256 + threadIdx.x;
  if (i < n) dst[i] = (_Float16)src[i];
}

// ---------------------------------------------------------------------------
// Kernel 1: QKV projection GEMM.  out[m][n] = sum_k x[m][k] * w[n][k] + b[n]
// M = 100352 (3136 blocks * 32 rows), K = 512, N = 1536.
// Block = 256 threads = 8 waves.  Each wave computes 32x32 macro-tiles
// (2x2 register tiling: 4 WMMA per 4 fragment loads per K-step).
// ---------------------------------------------------------------------------
#define XLD 520  // 260 dwords -> row*260 % 64 = row*4: conflict-free A reads
__global__ __launch_bounds__(256) void qkv_gemm(
    const float* __restrict__ x, const _Float16* __restrict__ w,
    const float* __restrict__ bias, _Float16* __restrict__ out) {
  __shared__ _Float16 xs[32 * XLD];
  const int tid = threadIdx.x;
  const int m0  = blockIdx.x * 32;
  for (int idx = tid; idx < 32 * 512; idx += 256) {
    const int r = idx >> 9, c = idx & 511;
    xs[r * XLD + c] = (_Float16)x[(size_t)(m0 + r) * 512 + c];
  }
  __syncthreads();
  const int wid = tid >> 5, lane = tid & 31;
  const int colL  = lane & 15;
  const int rbase = (lane < 16) ? 0 : 8;
  // wave owns n-tiles [wid*12, wid*12+12), processed as 6 pairs
  for (int np = 0; np < 6; ++np) {
    const int n0 = (wid * 12 + np * 2) * 16;
    v8f acc00 = {}, acc01 = {}, acc10 = {}, acc11 = {};
#pragma unroll 2
    for (int k0 = 0; k0 < 512; k0 += 32) {
      v16h a0 = frag_rowmajor(xs + k0, XLD, lane);
      v16h a1 = frag_rowmajor(xs + 16 * XLD + k0, XLD, lane);
      v16h b0 = frag_rowmajor(w + (size_t)n0 * 512 + k0, 512, lane);
      v16h b1 = frag_rowmajor(w + (size_t)(n0 + 16) * 512 + k0, 512, lane);
      acc00 = wmma_f16(a0, b0, acc00);
      acc01 = wmma_f16(a0, b1, acc01);
      acc10 = wmma_f16(a1, b0, acc10);
      acc11 = wmma_f16(a1, b1, acc11);
    }
    const float bv0 = bias[n0 + colL];
    const float bv1 = bias[n0 + 16 + colL];
#pragma unroll
    for (int r = 0; r < 8; ++r) {
      const size_t row0 = (size_t)(m0 + rbase + r) * 1536;
      const size_t row1 = (size_t)(m0 + 16 + rbase + r) * 1536;
      out[row0 + n0 + colL]      = (_Float16)(acc00[r] + bv0);
      out[row0 + n0 + 16 + colL] = (_Float16)(acc01[r] + bv1);
      out[row1 + n0 + colL]      = (_Float16)(acc10[r] + bv0);
      out[row1 + n0 + 16 + colL] = (_Float16)(acc11[r] + bv1);
    }
  }
}

// ---------------------------------------------------------------------------
// Kernel 2: fused window attention per (window b, head h).
// N = 98 padded to 112 rows / 128 K; hd = 32 (exactly one WMMA K-step for S).
// ---------------------------------------------------------------------------
#define QLD  40   // 20 dwords -> row*20 % 64 distinct for 16 rows
#define SLD  136  // 68 dwords -> row*4 bank stride
__global__ __launch_bounds__(256) void attn_kernel(
    const _Float16* __restrict__ qkv, const float* __restrict__ mask,
    const float* __restrict__ bias_table, const int* __restrict__ rpi,
    _Float16* __restrict__ aout) {
  __shared__ _Float16 qs[128 * QLD];
  __shared__ _Float16 ks2[128 * QLD];
  __shared__ _Float16 vs[128 * QLD];
  __shared__ _Float16 ss[112 * SLD];  // logits -> probabilities, 128 K cols

  const int b = blockIdx.x, h = blockIdx.y;
  const int tid = threadIdx.x;

  // Stage q/k/v (f16) with zero padding to 128 rows.
  for (int idx = tid; idx < 128 * 32; idx += 256) {
    const int n = idx >> 5, d = idx & 31;
    _Float16 qv = (_Float16)0.f, kv = (_Float16)0.f, vv = (_Float16)0.f;
    if (n < 98) {
      const size_t base = (size_t)(b * 98 + n) * 1536 + h * 32 + d;
      qv = qkv[base];
      kv = qkv[base + 512];
      vv = qkv[base + 1024];
    }
    qs[n * QLD + d]  = qv;
    ks2[n * QLD + d] = kv;
    vs[n * QLD + d]  = vv;
  }
  // Pre-fill logits with a large negative so padded columns soft-max to 0.
  for (int idx = tid; idx < 112 * SLD; idx += 256) ss[idx] = (_Float16)(-30000.f);
  __syncthreads();

  const int wid = tid >> 5, lane = tid & 31;
  const int colL  = lane & 15;
  const int rbase = (lane < 16) ? 0 : 8;
  const float scale = 0.17677669529663687f;  // 32^-0.5
  const float* maskb = mask + (size_t)(b & 63) * 98 * 98;

  // S = Q K^T  (49 tiles of 16x16; K=32=hd -> single WMMA per tile)
  for (int tile = wid; tile < 49; tile += 8) {
    const int tr = tile / 7, tc = tile % 7;
    v16h a = frag_rowmajor(qs + tr * 16 * QLD, QLD, lane);
    v16h kb = frag_rowmajor(ks2 + tc * 16 * QLD, QLD, lane);  // B[k=d][col=m]
    v8f acc = {};
    acc = wmma_f16(a, kb, acc);
    const int m = tc * 16 + colL;
#pragma unroll
    for (int r = 0; r < 8; ++r) {
      const int n = tr * 16 + rbase + r;
      float val = -30000.f;
      if (n < 98 && m < 98)
        val = acc[r] * scale + bias_table[rpi[n * 98 + m] * 16 + h] +
              maskb[n * 98 + m];
      ss[n * SLD + m] = (_Float16)val;
    }
  }
  __syncthreads();

  // Row softmax over 128 (padded) columns.
  for (int r = tid; r < 112; r += 256) {
    if (r < 98) {
      float mx = -3.0e30f;
      for (int m = 0; m < 128; ++m) mx = fmaxf(mx, (float)ss[r * SLD + m]);
      float sum = 0.f;
      for (int m = 0; m < 128; ++m) {
        const float e = __expf((float)ss[r * SLD + m] - mx);
        sum += e;
        ss[r * SLD + m] = (_Float16)e;
      }
      const float inv = 1.0f / sum;
      for (int m = 0; m < 128; ++m)
        ss[r * SLD + m] = (_Float16)((float)ss[r * SLD + m] * inv);
    } else {
      for (int m = 0; m < 128; ++m) ss[r * SLD + m] = (_Float16)0.f;
    }
  }
  __syncthreads();

  // O = P V   (7 row-tiles x 2 col-tiles, 4 K-steps of 32)
  for (int tile = wid; tile < 14; tile += 8) {
    const int tr = tile >> 1;
    const int d0 = (tile & 1) * 16;
    v8f acc = {};
#pragma unroll
    for (int kt = 0; kt < 4; ++kt) {
      v16h a = frag_rowmajor(ss + tr * 16 * SLD + kt * 32, SLD, lane);
      v16h vb = frag_kmajor(vs + kt * 32 * QLD + d0, QLD, lane);  // B[k][col=d]
      acc = wmma_f16(a, vb, acc);
    }
#pragma unroll
    for (int r = 0; r < 8; ++r) {
      const int n = tr * 16 + rbase + r;
      if (n < 98)
        aout[(size_t)(b * 98 + n) * 512 + h * 32 + d0 + colL] = (_Float16)acc[r];
    }
  }
}

// ---------------------------------------------------------------------------
// Kernel 3: output projection GEMM + bias -> fp32 result.
// out[m][n] = sum_k a[m][k] * w[n][k] + b[n];  M=100352, K=N=512.
// 32-row blocks, 2x2 register tiling per wave (wave owns 4 n-tiles).
// ---------------------------------------------------------------------------
__global__ __launch_bounds__(256) void proj_gemm(
    const _Float16* __restrict__ a, const _Float16* __restrict__ w,
    const float* __restrict__ bias, float* __restrict__ out) {
  __shared__ _Float16 as[32 * XLD];
  const int tid = threadIdx.x;
  const int m0  = blockIdx.x * 32;
  for (int idx = tid; idx < 32 * 512; idx += 256) {
    const int r = idx >> 9, c = idx & 511;
    as[r * XLD + c] = a[(size_t)(m0 + r) * 512 + c];
  }
  __syncthreads();
  const int wid = tid >> 5, lane = tid & 31;
  const int colL  = lane & 15;
  const int rbase = (lane < 16) ? 0 : 8;
  for (int np = 0; np < 2; ++np) {
    const int n0 = (wid * 4 + np * 2) * 16;
    v8f acc00 = {}, acc01 = {}, acc10 = {}, acc11 = {};
#pragma unroll 2
    for (int k0 = 0; k0 < 512; k0 += 32) {
      v16h a0 = frag_rowmajor(as + k0, XLD, lane);
      v16h a1 = frag_rowmajor(as + 16 * XLD + k0, XLD, lane);
      v16h b0 = frag_rowmajor(w + (size_t)n0 * 512 + k0, 512, lane);
      v16h b1 = frag_rowmajor(w + (size_t)(n0 + 16) * 512 + k0, 512, lane);
      acc00 = wmma_f16(a0, b0, acc00);
      acc01 = wmma_f16(a0, b1, acc01);
      acc10 = wmma_f16(a1, b0, acc10);
      acc11 = wmma_f16(a1, b1, acc11);
    }
    const float bv0 = bias[n0 + colL];
    const float bv1 = bias[n0 + 16 + colL];
#pragma unroll
    for (int r = 0; r < 8; ++r) {
      const size_t row0 = (size_t)(m0 + rbase + r) * 512;
      const size_t row1 = (size_t)(m0 + 16 + rbase + r) * 512;
      out[row0 + n0 + colL]      = acc00[r] + bv0;
      out[row0 + n0 + 16 + colL] = acc01[r] + bv1;
      out[row1 + n0 + colL]      = acc10[r] + bv0;
      out[row1 + n0 + 16 + colL] = acc11[r] + bv1;
    }
  }
}

// ---------------------------------------------------------------------------
// Host-side orchestration
// ---------------------------------------------------------------------------
extern "C" void kernel_launch(void* const* d_in, const int* in_sizes, int n_in,
                              void* d_out, int out_size, void* d_ws, size_t ws_size,
                              hipStream_t stream) {
  const float* x          = (const float*)d_in[0];
  const float* mask       = (const float*)d_in[1];
  const float* qkv_w      = (const float*)d_in[2];
  const float* qkv_b      = (const float*)d_in[3];
  const float* proj_w     = (const float*)d_in[4];
  const float* proj_b     = (const float*)d_in[5];
  const float* bias_table = (const float*)d_in[6];
  const int*   rpi        = (const int*)d_in[7];

  char* ws = (char*)d_ws;
  // M = 1024*98 = 100352 rows
  _Float16* qkv16  = (_Float16*)ws;                          // 100352*1536*2 = 308,281,344 B
  _Float16* attn16 = (_Float16*)(ws + 308281344u);           // 100352*512*2  = 102,760,448 B
  _Float16* wq16   = (_Float16*)(ws + 411041792u);           // 1536*512*2    =   1,572,864 B
  _Float16* wp16   = (_Float16*)(ws + 412614656u);           // 512*512*2     =     524,288 B

  cvt_f32_f16<<<(1536 * 512 + 255) / 256, 256, 0, stream>>>(qkv_w, wq16, 1536 * 512);
  cvt_f32_f16<<<(512 * 512 + 255) / 256, 256, 0, stream>>>(proj_w, wp16, 512 * 512);

  qkv_gemm<<<3136, 256, 0, stream>>>(x, wq16, qkv_b, qkv16);
  attn_kernel<<<dim3(1024, 16), 256, 0, stream>>>(qkv16, mask, bias_table, rpi, attn16);
  proj_gemm<<<3136, 256, 0, stream>>>(attn16, wp16, proj_b, (float*)d_out);
}
